// GenerativeODE_7713761264214
// MI455X (gfx1250) — matrix operand order repaired and checked by reading, not compiled, not run on hardware
//
#include <hip/hip_runtime.h>
#include <hip/hip_bf16.h>
#include <math.h>

typedef __attribute__((ext_vector_type(16))) _Float16 v16h;
typedef __attribute__((ext_vector_type(8)))  _Float16 v8h;
typedef __attribute__((ext_vector_type(8)))  float    v8f;
typedef __attribute__((ext_vector_type(4)))  float    f4;

namespace {
constexpr int B_  = 1024, T_ = 48, Zn = 2000, PFn = 64, ZEn = 8, Hn = 128;
constexpr int SDn = 137, SDP = 144;
constexpr float ATTc = 0.1f;
constexpr float W24  = 0.26179938779914943f; // 2*pi/24

// padded, fragment-packed f16 weight offsets inside workspace (units: halves)
constexpr int OFF_WIN  = 0;        // 256 x 160
constexpr int OFF_WA0  = 40960;    // 256 x 256
constexpr int OFF_WB0  = 106496;   // 256 x 256 (contiguous after WA0)
constexpr int OFF_WA1  = 172032;   // 256 x 256
constexpr int OFF_WB1  = 237568;   // 256 x 256
constexpr int OFF_WOUT = 303104;   // 144 x 256
constexpr int OFF_WV   = 339968;   // 128 x 128
constexpr int OFF_WO   = 356352;   // 128 x 128
constexpr int OFF_ENC1 = 372736;   // 256 x 96
constexpr int OFF_ENC2 = 397312;   // 256 x 256

// float region of workspace (units: floats, region starts at byte 1 MiB)
constexpr size_t FO_HOME = 262144;   // B*8
constexpr size_t FO_WORK = 270336;   // B*8
constexpr size_t FO_CAND = 278528;   // Z*8 (padded)
constexpr size_t FO_TGT  = 294912;   // B*T*8
constexpr size_t FO_PRED = 688128;   // B*T*137

// output offsets (floats) in d_out, reference return order
constexpr size_t OUT_PATH = 98304000ull;
constexpr size_t OUT_PURP = 98697216ull;
constexpr size_t OUT_MODE = 98942976ull;
constexpr size_t OUT_MU   = 99139584ull;
constexpr size_t OUT_LV   = 99270656ull;

// bias LDS layout (floats)
constexpr int BL_IN   = 0;     // 256
constexpr int BL_BA   = 256;   // 512 (ba0 | ba1)
constexpr int BL_BB   = 768;   // 512 (bb0 | bb1)
constexpr int BL_OUT  = 1280;  // 144 (zero padded past 137)
constexpr int BL_ATV  = 1424;  // 128
constexpr int BL_ATO  = 1552;  // 128
constexpr int BL_ALWV = 1680;  // 64
constexpr int BL_ALBV = 1744;  // 8
constexpr int BL_ALWO = 1752;  // 64
constexpr int BL_ALBO = 1816;  // 8
constexpr int BL_TOT  = 1824;
} // namespace

__device__ __forceinline__ float fast_tanh(float x) {
#if __has_builtin(__builtin_amdgcn_tanhf)
  return __builtin_amdgcn_tanhf(x);   // native V_TANH_F32 on CDNA5
#else
  return tanhf(x);
#endif
}

// ---------------------------------------------------------------------------
// Workgroup-cooperative GEMM:  D[16 x N] = A[16 x K] @ W^T.
// W is fragment-packed: 512-half tiles indexed (nt*nkt + kt); within a tile a
// lane's 16 B-fragment halves are contiguous -> two 16-byte loads (works for
// both global and LDS-resident W).  A fragments come from LDS as two 16-byte
// loads (16-bit A 16x32 layout, ISA 7.12.2).
// C/D layout: comp g -> m = g + 8*(lane>>4), n = lane&15.
// ---------------------------------------------------------------------------
template <bool AHALF, class EPI>
__device__ __forceinline__ void wg_gemm(const void* Aptr, int lda,
                                        const _Float16* Wpacked,
                                        int K, int N, int lane, int wave,
                                        EPI epi) {
  const int am  = lane & 15;
  const int kg  = (lane >> 4) * 8;
  const int nkt = K >> 5;
  for (int nt = wave; nt * 16 < N; nt += 8) {
    v8f acc = {};
    const _Float16* Wtile = Wpacked + (((size_t)nt * nkt) << 9) + (lane << 4);
    for (int kt = 0; kt < nkt; ++kt) {
      v16h afrag;
      if constexpr (AHALF) {
        const _Float16* Arow =
            (const _Float16*)Aptr + (size_t)am * lda + (kt << 5);
        v8h lo = *(const v8h*)(Arow + kg);
        v8h hi = *(const v8h*)(Arow + 16 + kg);
        afrag = __builtin_shufflevector(lo, hi, 0, 1, 2, 3, 4, 5, 6, 7, 8, 9,
                                        10, 11, 12, 13, 14, 15);
      } else {
        const float* Arow = (const float*)Aptr + (size_t)am * lda + (kt << 5);
        f4 a0 = *(const f4*)(Arow + kg);
        f4 a1 = *(const f4*)(Arow + kg + 4);
        f4 a2 = *(const f4*)(Arow + 16 + kg);
        f4 a3 = *(const f4*)(Arow + 16 + kg + 4);
#pragma unroll
        for (int j = 0; j < 4; ++j) {
          afrag[j]      = (_Float16)a0[j];
          afrag[4 + j]  = (_Float16)a1[j];
          afrag[8 + j]  = (_Float16)a2[j];
          afrag[12 + j] = (_Float16)a3[j];
        }
      }
      v8h blo = *(const v8h*)(Wtile);
      v8h bhi = *(const v8h*)(Wtile + 8);
      v16h bfrag = __builtin_shufflevector(blo, bhi, 0, 1, 2, 3, 4, 5, 6, 7, 8,
                                           9, 10, 11, 12, 13, 14, 15);
      Wtile += 512;
      acc = __builtin_amdgcn_wmma_f32_16x16x32_f16(false, afrag, false, bfrag,
                                                   (short)0, acc, false, false);
    }
    const int n  = nt * 16 + (lane & 15);
    const int mb = (lane >> 4) * 8;
#pragma unroll
    for (int g = 0; g < 8; ++g) epi(mb + g, n, acc[g]);
  }
}

// ---------------------------------------------------------------------------
// weight convert + pad + fragment-pack
// ---------------------------------------------------------------------------
__global__ __launch_bounds__(256) void cvt_kernel(const float* __restrict__ src,
                                                  _Float16* __restrict__ dst,
                                                  int N, int K, int Npad,
                                                  int Kpad) {
  int idx = blockIdx.x * 256 + threadIdx.x;
  if (idx >= Npad * Kpad) return;
  int tile = idx >> 9, w = idx & 511;
  int lane = w >> 4, j = w & 15;
  int nkt = Kpad >> 5;
  int n = (tile / nkt) * 16 + (lane & 15);
  int k = (tile % nkt) * 32 + ((lane >> 4) << 3) + ((j >> 3) << 4) + (j & 7);
  dst[idx] = (n < N && k < K) ? (_Float16)src[(size_t)n * K + k] : (_Float16)0.f;
}

// ---------------------------------------------------------------------------
// zone-feature embeddings
// ---------------------------------------------------------------------------
__global__ __launch_bounds__(256) void zf_kernel(
    const float* __restrict__ azf, const float* __restrict__ hzf,
    const float* __restrict__ wzf, const float* __restrict__ zfW,
    const float* __restrict__ zfb, float* __restrict__ cand,
    float* __restrict__ home_e, float* __restrict__ work_e) {
  int idx = blockIdx.x * 256 + threadIdx.x;
  int total = (Zn + 2 * B_) * 8;
  if (idx >= total) return;
  int row = idx >> 3, j = idx & 7;
  const float* src;
  float* dst;
  if (row < Zn) {
    src = azf + (size_t)row * 32;
    dst = cand + (size_t)row * 8;
  } else if (row < Zn + B_) {
    src = hzf + (size_t)(row - Zn) * 32;
    dst = home_e + (size_t)(row - Zn) * 8;
  } else {
    src = wzf + (size_t)(row - Zn - B_) * 32;
    dst = work_e + (size_t)(row - Zn - B_) * 8;
  }
  float acc = zfb[j];
#pragma unroll
  for (int k = 0; k < 32; ++k) acc += zfW[j * 32 + k] * src[k];
  dst[j] = acc;
}

// ---------------------------------------------------------------------------
// encoder
// ---------------------------------------------------------------------------
__global__ __launch_bounds__(256) void enc_kernel(
    const float* __restrict__ pf, const float* __restrict__ home_e,
    const float* __restrict__ work_e, const float* __restrict__ eps,
    const _Float16* __restrict__ enc1W, const float* __restrict__ enc1b,
    const _Float16* __restrict__ enc2W, const float* __restrict__ enc2b,
    float* __restrict__ pred_s, float* __restrict__ out) {
  __shared__ alignas(16) float    Ain[16 * 96];
  __shared__ alignas(16) _Float16 Xh[16 * 256];
  __shared__ alignas(16) float    Hp[16 * 256];
  int tid = threadIdx.x, lane = tid & 31, wave = tid >> 5;
  int b0 = blockIdx.x * 16;
  for (int idx = tid; idx < 16 * 96; idx += 256) {
    int m = idx / 96, c = idx % 96, b = b0 + m;
    float v;
    if (c < 64)      v = pf[(size_t)b * PFn + c];
    else if (c < 72) v = home_e[(size_t)b * 8 + (c - 64)];
    else if (c < 80) v = work_e[(size_t)b * 8 + (c - 72)];
    else             v = 0.f;
    Ain[idx] = v;
  }
  __syncthreads();
  wg_gemm<false>(Ain, 96, enc1W, 96, 256, lane, wave,
                 [&](int m, int n, float a) {
                   float v = a + enc1b[n];
                   Xh[m * 256 + n] = (_Float16)(v > 0.f ? v : 0.f);
                 });
  __syncthreads();
  wg_gemm<true>(Xh, 256, enc2W, 256, 256, lane, wave,
                [&](int m, int n, float a) { Hp[m * 256 + n] = a + enc2b[n]; });
  __syncthreads();
  for (int idx = tid; idx < 16 * Hn; idx += 256) {
    int m = idx / Hn, j = idx % Hn, b = b0 + m;
    float mu = Hp[m * 256 + j], lv = Hp[m * 256 + 128 + j];
    out[OUT_MU + (size_t)b * Hn + j] = mu;
    out[OUT_LV + (size_t)b * Hn + j] = lv;
    float h0 = mu + expf(0.5f * lv) * eps[(size_t)b * Hn + j];
    pred_s[((size_t)b * T_) * SDn + j] = h0;
  }
  for (int idx = tid; idx < 16 * 9; idx += 256) {
    int m = idx / 9, c = idx % 9, b = b0 + m;
    pred_s[((size_t)b * T_) * SDn + Hn + c] =
        (c < 8) ? home_e[(size_t)b * 8 + c] : 0.f;
  }
}

// ---------------------------------------------------------------------------
// one _ode_f evaluation on a 16-row tile; layer-0 residual weights come from
// the LDS weight cache, the rest stream from L2
// ---------------------------------------------------------------------------
__device__ __forceinline__ void ode_f_eval(float* ev, _Float16* Xh,
                                           _Float16* Yh, float* Fb, float* vloc,
                                           const _Float16* wcache,
                                           const _Float16* wsh, const float* bl,
                                           int tid, int lane, int wave) {
  // x = relu([s, tv] @ Win^T + b_in)      (K padded 139 -> 160)
  wg_gemm<false>(ev, 160, wsh + OFF_WIN, 160, 256, lane, wave,
                 [&](int m, int n, float a) {
                   float v = a + bl[BL_IN + n];
                   Xh[m * 256 + n] = (_Float16)(v > 0.f ? v : 0.f);
                 });
  __syncthreads();
  // residual block 0 (LDS-cached weights)
  wg_gemm<true>(Xh, 256, wcache, 256, 256, lane, wave,
                [&](int m, int n, float a) {
                  Yh[m * 256 + n] = (_Float16)fast_tanh(a + bl[BL_BA + n]);
                });
  __syncthreads();
  wg_gemm<true>(Yh, 256, wcache + 65536, 256, 256, lane, wave,
                [&](int m, int n, float a) {
                  float x = (float)Xh[m * 256 + n];
                  Xh[m * 256 + n] = (_Float16)fast_tanh(x + a + bl[BL_BB + n]);
                });
  __syncthreads();
  // residual block 1 (streamed from L2)
  wg_gemm<true>(Xh, 256, wsh + OFF_WA1, 256, 256, lane, wave,
                [&](int m, int n, float a) {
                  Yh[m * 256 + n] =
                      (_Float16)fast_tanh(a + bl[BL_BA + 256 + n]);
                });
  __syncthreads();
  wg_gemm<true>(Yh, 256, wsh + OFF_WB1, 256, 256, lane, wave,
                [&](int m, int n, float a) {
                  float x = (float)Xh[m * 256 + n];
                  Xh[m * 256 + n] =
                      (_Float16)fast_tanh(x + a + bl[BL_BB + 256 + n]);
                });
  __syncthreads();
  // core -> Fb (N padded 137 -> 144; BL_OUT zero padded)
  wg_gemm<true>(Xh, 256, wsh + OFF_WOUT, 256, 144, lane, wave,
                [&](int m, int n, float a) {
                  Fb[m * SDP + n] = a + bl[BL_OUT + n];
                });
  __syncthreads();
  // d_h stage 1: Yh[:, :128] = h @ Wv^T + bv   (h = ev[:, :128])
  wg_gemm<false>(ev, 160, wsh + OFF_WV, 128, 128, lane, wave,
                 [&](int m, int n, float a) {
                   Yh[m * 128 + n] = (_Float16)(a + bl[BL_ATV + n]);
                 });
  __syncthreads();
  // d_h stage 2: F[:, :128] += ATT * ((z1 @ Wo^T + bo) - h)
  wg_gemm<true>(Yh, 128, wsh + OFF_WO, 128, 128, lane, wave,
                [&](int m, int n, float a) {
                  Fb[m * SDP + n] +=
                      ATTc * ((a + bl[BL_ATO + n]) - ev[m * 160 + n]);
                });
  __syncthreads();
  // d_loc: two 8x8 linears on loc = ev[:, 128:136]
  if (tid < 128) {
    int m = tid >> 3, j = tid & 7;
    float acc = bl[BL_ALBV + j];
#pragma unroll
    for (int k = 0; k < 8; ++k)
      acc += bl[BL_ALWV + j * 8 + k] * ev[m * 160 + 128 + k];
    vloc[m * 8 + j] = acc;
  }
  __syncthreads();
  if (tid < 128) {
    int m = tid >> 3, j = tid & 7;
    float acc = bl[BL_ALBO + j];
#pragma unroll
    for (int k = 0; k < 8; ++k) acc += bl[BL_ALWO + j * 8 + k] * vloc[m * 8 + k];
    Fb[m * SDP + 128 + j] += ATTc * (acc - ev[m * 160 + 128 + j]);
  }
  __syncthreads();
}

// ---------------------------------------------------------------------------
// persistent RK4 integrator: one workgroup per 16 batch rows, 47 steps.
// State (s, k-sum) lives in per-thread registers over a fixed 16x144 stripe;
// layer-0 residual weights (256 KB) are cached in dynamic LDS for the whole
// kernel (ISA: up to 320 KB LDS per workgroup).
// ---------------------------------------------------------------------------
__global__ __launch_bounds__(256) void ode_kernel(
    const float* __restrict__ times, const _Float16* __restrict__ wsh,
    const float* __restrict__ b_in, const float* __restrict__ res_ba,
    const float* __restrict__ res_bb, const float* __restrict__ b_out,
    const float* __restrict__ al_Wv, const float* __restrict__ al_bv,
    const float* __restrict__ al_Wo, const float* __restrict__ al_bo,
    const float* __restrict__ at_bv, const float* __restrict__ at_bo,
    float* __restrict__ pred_s) {
  __shared__ alignas(16) float    Fb[16 * SDP];
  __shared__ alignas(16) float    ev[16 * 160];
  __shared__ alignas(16) _Float16 Xh[16 * 256];
  __shared__ alignas(16) _Float16 Yh[16 * 256];
  __shared__ alignas(16) float    vloc[16 * 8];
  __shared__ alignas(16) float    bl[BL_TOT];
  extern __shared__ _Float16 wlds[];  // 262144 B: packed Wa0 | Wb0

  int tid = threadIdx.x, lane = tid & 31, wave = tid >> 5;
  int b0 = blockIdx.x * 16;

  // preload biases + tiny al_* weights into LDS
  for (int i = tid; i < BL_TOT; i += 256) {
    float v;
    if (i < 256)       v = b_in[i];
    else if (i < 768)  v = res_ba[i - 256];
    else if (i < 1280) v = res_bb[i - 768];
    else if (i < 1424) { int c = i - 1280; v = (c < SDn) ? b_out[c] : 0.f; }
    else if (i < 1552) v = at_bv[i - 1424];
    else if (i < 1680) v = at_bo[i - 1552];
    else if (i < 1744) v = al_Wv[i - 1680];
    else if (i < 1752) v = al_bv[i - 1744];
    else if (i < 1816) v = al_Wo[i - 1752];
    else               v = al_bo[i - 1816];
    bl[i] = v;
  }
  // cache layer-0 residual weights (contiguous packed Wa0|Wb0) into LDS
  {
    const f4* src = (const f4*)(wsh + OFF_WA0);
    f4* dst = (f4*)wlds;
    for (int i = tid; i < 262144 / 16; i += 256) dst[i] = src[i];
  }
  // per-thread state stripes over the 16 x SDP domain (9 elems each)
  float s_reg[9], k_reg[9];
#pragma unroll
  for (int i = 0; i < 9; ++i) {
    int idx = tid + 256 * i;
    int m = idx / SDP, col = idx % SDP;
    s_reg[i] =
        (col < SDn) ? pred_s[((size_t)(b0 + m) * T_) * SDn + col] : 0.f;
  }
  for (int idx = tid; idx < 16 * SDP; idx += 256) Fb[idx] = 0.f;
  __syncthreads();

  for (int t = 0; t < T_ - 1; ++t) {
    float t0 = times[t], t1 = times[t + 1], dt = t1 - t0;
#pragma unroll
    for (int i = 0; i < 9; ++i) k_reg[i] = 0.f;
    for (int s = 0; s < 4; ++s) {
      float c = (s == 0) ? 0.f : ((s == 3) ? dt : 0.5f * dt);
      float w = (s == 1 || s == 2) ? 2.f : 1.f;
      float te = t0 + c;
      float tvs = __sinf(te * W24), tvc = __cosf(te * W24);
      // build eval state (uses previous stage's Fb; c==0 for s==0)
#pragma unroll
      for (int i = 0; i < 9; ++i) {
        int idx = tid + 256 * i;
        int m = idx / SDP, col = idx % SDP;
        if (col < SDn) ev[m * 160 + col] = s_reg[i] + c * Fb[idx];
      }
      for (int q = tid; q < 16 * 23; q += 256) {
        int m = q / 23, col = 137 + q % 23;
        ev[m * 160 + col] = (col == 137) ? tvs : ((col == 138) ? tvc : 0.f);
      }
      __syncthreads();
      ode_f_eval(ev, Xh, Yh, Fb, vloc, wlds, wsh, bl, tid, lane, wave);
#pragma unroll
      for (int i = 0; i < 9; ++i) k_reg[i] += w * Fb[tid + 256 * i];
    }
#pragma unroll
    for (int i = 0; i < 9; ++i) {
      int idx = tid + 256 * i;
      int m = idx / SDP, col = idx % SDP;
      float v = s_reg[i] + (dt / 6.f) * k_reg[i];
      s_reg[i] = v;
      if (col < SDn) pred_s[((size_t)(b0 + m) * T_ + (t + 1)) * SDn + col] = v;
    }
  }
}

// ---------------------------------------------------------------------------
// decode heads: loc_path, tgt, purp/mode logits (single pass over hidden)
// ---------------------------------------------------------------------------
__global__ __launch_bounds__(256) void heads_kernel(
    const float* __restrict__ pred_s, const float* __restrict__ decW,
    const float* __restrict__ decb, const float* __restrict__ movW,
    const float* __restrict__ movb, const float* __restrict__ stayW,
    const float* __restrict__ stayb, const float* __restrict__ modeW,
    const float* __restrict__ modeb, float* __restrict__ tgt,
    float* __restrict__ out) {
  int row = blockIdx.x * 256 + threadIdx.x;
  if (row >= B_ * T_) return;
  const float* s = pred_s + (size_t)row * SDn;
#pragma unroll
  for (int j = 0; j < 8; ++j) out[OUT_PATH + (size_t)row * 8 + j] = s[Hn + j];

  float acc[16];
#pragma unroll
  for (int j = 0; j < 8; ++j) acc[j] = decb[j];
  acc[8] = movb[0];
#pragma unroll
  for (int j = 0; j < 4; ++j) acc[9 + j] = stayb[j];
#pragma unroll
  for (int j = 0; j < 3; ++j) acc[13 + j] = modeb[j];
  for (int k = 0; k < Hn; ++k) {
    float x = s[k];
#pragma unroll
    for (int j = 0; j < 8; ++j) acc[j] += decW[j * Hn + k] * x;
    acc[8] += movW[k] * x;
#pragma unroll
    for (int j = 0; j < 4; ++j) acc[9 + j] += stayW[j * Hn + k] * x;
#pragma unroll
    for (int j = 0; j < 3; ++j) acc[13 + j] += modeW[j * Hn + k] * x;
  }
#pragma unroll
  for (int j = 0; j < 8; ++j) tgt[(size_t)row * 8 + j] = acc[j];
  float ml = acc[8];
  float mv = 1.f / (1.f + expf(-ml));
#pragma unroll
  for (int j = 0; j < 4; ++j)
    out[OUT_PURP + (size_t)row * 5 + j] = acc[9 + j] * (1.f - mv);
  out[OUT_PURP + (size_t)row * 5 + 4] = ml;
  out[OUT_MODE + (size_t)row * 4 + 0] = -ml;
#pragma unroll
  for (int j = 0; j < 3; ++j)
    out[OUT_MODE + (size_t)row * 4 + 1 + j] = acc[13 + j] * mv;
}

// ---------------------------------------------------------------------------
// loc_logits = tgt @ cand^T  (pure bandwidth: 393 MB of stores, float4 wide)
// ---------------------------------------------------------------------------
__global__ __launch_bounds__(256) void logits_kernel(
    const float* __restrict__ tgt, const float* __restrict__ cand,
    float* __restrict__ out) {
  int bt = blockIdx.x;  // uniform per block -> scalarized tgt loads
  int z0 = (blockIdx.y * 256 + threadIdx.x) * 4;
  if (z0 >= Zn) return;
  float t[8];
#pragma unroll
  for (int j = 0; j < 8; ++j) t[j] = tgt[(size_t)bt * 8 + j];
  f4 r;
#pragma unroll
  for (int q = 0; q < 4; ++q) {
    f4 c0 = *(const f4*)(cand + (size_t)(z0 + q) * 8);
    f4 c1 = *(const f4*)(cand + (size_t)(z0 + q) * 8 + 4);
    r[q] = t[0] * c0[0] + t[1] * c0[1] + t[2] * c0[2] + t[3] * c0[3] +
           t[4] * c1[0] + t[5] * c1[1] + t[6] * c1[2] + t[7] * c1[3];
  }
  *(f4*)(out + (size_t)bt * Zn + z0) = r;
}

// ---------------------------------------------------------------------------
extern "C" void kernel_launch(void* const* d_in, const int* in_sizes, int n_in,
                              void* d_out, int out_size, void* d_ws,
                              size_t ws_size, hipStream_t stream) {
  (void)in_sizes; (void)n_in; (void)out_size; (void)ws_size;
  const float* pf      = (const float*)d_in[0];
  const float* hzf     = (const float*)d_in[1];
  const float* wzf     = (const float*)d_in[2];
  const float* times   = (const float*)d_in[4];
  const float* azf     = (const float*)d_in[5];
  const float* eps     = (const float*)d_in[7];
  const float* zf_W    = (const float*)d_in[8];
  const float* zf_b    = (const float*)d_in[9];
  const float* enc1_W  = (const float*)d_in[10];
  const float* enc1_b  = (const float*)d_in[11];
  const float* enc2_W  = (const float*)d_in[12];
  const float* enc2_b  = (const float*)d_in[13];
  const float* odein_W = (const float*)d_in[14];
  const float* odein_b = (const float*)d_in[15];
  const float* res_Wa  = (const float*)d_in[16];
  const float* res_ba  = (const float*)d_in[17];
  const float* res_Wb  = (const float*)d_in[18];
  const float* res_bb  = (const float*)d_in[19];
  const float* odeout_W= (const float*)d_in[20];
  const float* odeout_b= (const float*)d_in[21];
  const float* al_Wv   = (const float*)d_in[22];
  const float* al_bv   = (const float*)d_in[23];
  const float* al_Wo   = (const float*)d_in[24];
  const float* al_bo   = (const float*)d_in[25];
  const float* at_Wv   = (const float*)d_in[26];
  const float* at_bv   = (const float*)d_in[27];
  const float* at_Wo   = (const float*)d_in[28];
  const float* at_bo   = (const float*)d_in[29];
  const float* dec_W   = (const float*)d_in[30];
  const float* dec_b   = (const float*)d_in[31];
  const float* mov_W   = (const float*)d_in[32];
  const float* mov_b   = (const float*)d_in[33];
  const float* stay_W  = (const float*)d_in[34];
  const float* stay_b  = (const float*)d_in[35];
  const float* mode_W  = (const float*)d_in[36];
  const float* mode_b  = (const float*)d_in[37];

  _Float16* wsh = (_Float16*)d_ws;
  float*    wsf = (float*)d_ws;
  float*    out = (float*)d_out;
  float*    pred_s = wsf + FO_PRED;

  // 1) convert/pad/fragment-pack all GEMM weights to f16 in workspace
  struct Job { const float* src; int dst; int N, K, Np, Kp; };
  const Job jobs[] = {
      {odein_W,        OFF_WIN,  256, 139, 256, 160},
      {res_Wa,         OFF_WA0,  256, 256, 256, 256},
      {res_Wa + 65536, OFF_WA1,  256, 256, 256, 256},
      {res_Wb,         OFF_WB0,  256, 256, 256, 256},
      {res_Wb + 65536, OFF_WB1,  256, 256, 256, 256},
      {odeout_W,       OFF_WOUT, 137, 256, 144, 256},
      {at_Wv,          OFF_WV,   128, 128, 128, 128},
      {at_Wo,          OFF_WO,   128, 128, 128, 128},
      {enc1_W,         OFF_ENC1, 256,  80, 256,  96},
      {enc2_W,         OFF_ENC2, 256, 256, 256, 256},
  };
  for (const Job& j : jobs) {
    int elems = j.Np * j.Kp;
    cvt_kernel<<<(elems + 255) / 256, 256, 0, stream>>>(j.src, wsh + j.dst,
                                                        j.N, j.K, j.Np, j.Kp);
  }

  // 2) zone embeddings
  {
    int total = (Zn + 2 * B_) * 8;
    zf_kernel<<<(total + 255) / 256, 256, 0, stream>>>(
        azf, hzf, wzf, zf_W, zf_b, wsf + FO_CAND, wsf + FO_HOME, wsf + FO_WORK);
  }

  // 3) encoder -> mu, log_var, s0
  enc_kernel<<<B_ / 16, 256, 0, stream>>>(pf, wsf + FO_HOME, wsf + FO_WORK, eps,
                                          wsh + OFF_ENC1, enc1_b,
                                          wsh + OFF_ENC2, enc2_b, pred_s, out);

  // 4) persistent RK4 ODE integration (WMMA + 256 KB LDS weight cache)
  ode_kernel<<<B_ / 16, 256, 262144, stream>>>(
      times, wsh, odein_b, res_ba, res_bb, odeout_b, al_Wv, al_bv, al_Wo,
      al_bo, at_bv, at_bo, pred_s);

  // 5) decode heads (loc_path, tgt, purp/mode logits)
  heads_kernel<<<(B_ * T_ + 255) / 256, 256, 0, stream>>>(
      pred_s, dec_W, dec_b, mov_W, mov_b, stay_W, stay_b, mode_W, mode_b,
      wsf + FO_TGT, out);

  // 6) loc_logits (bandwidth bound: 393 MB store)
  {
    dim3 grid(B_ * T_, (Zn + 1023) / 1024);
    logits_kernel<<<grid, 256, 0, stream>>>(wsf + FO_TGT, wsf + FO_CAND, out);
  }
}
